// AttentionHead_13030930776624
// MI455X (gfx1250) — matrix operand order repaired and checked by reading, not compiled, not run on hardware
//
#include <hip/hip_runtime.h>

// ---------------------------------------------------------------------------
// AttentionHead for MI455X (gfx1250, wave32, WMMA).
//   inputs: x (4,4096,1024) f32, mask (tril, ignored -> causal), W_Q/W_K/W_V (1024,64) f32
//   output: (4,4096,64) f32
//
// Pipeline:
//   wprep_kernel   : W fp32 -> bf16 pre-swizzled into WMMA B-fragment layout.
//   qkv_rope_kernel: Q/K/V projection with v_wmma_f32_16x16x32_bf16, RoPE in fp32
//                    accumulators; results stored bf16 already swizzled into
//                    A-fragment layout (Q) / B-fragment layout (K, V).
//   attn_kernel    : causal flash attention; K/V tiles staged cooperatively in
//                    LDS via GLOBAL_LOAD_ASYNC_TO_LDS_B128 (ASYNCcnt) when the
//                    builtin exists, double-buffered, 1 barrier/step; interior
//                    steps use a mask-free softmax; fully-masked steps skipped
//                    (wave-uniform branches, barriers stay uniform).
// ---------------------------------------------------------------------------

typedef __attribute__((ext_vector_type(16))) __bf16 v16bf;
typedef __attribute__((ext_vector_type(8)))  __bf16 v8bf;
typedef __attribute__((ext_vector_type(8)))  float  v8f;
typedef __attribute__((ext_vector_type(4)))  int    v4i;

#define BATCH 4
#define SEQ   4096
#define EMB   1024
#define DH    64
#define NROW  (BATCH * SEQ)       // 16384 global rows
#define WMAT  (EMB * DH)          // 65536 elements per weight matrix

#if defined(__HIP_DEVICE_COMPILE__) && __has_builtin(__builtin_amdgcn_global_load_async_to_lds_b128)
#define USE_ASYNC_LDS 1
#endif

struct bfpair { v8bf lo, hi; };

// Assemble a 16-element bf16 fragment from 32 contiguous bytes (2 x b128).
__device__ __forceinline__ v16bf load_frag(const __bf16* p) {
    bfpair t;
    t.lo = *(const v8bf*)(p);
    t.hi = *(const v8bf*)(p + 8);
    return __builtin_bit_cast(v16bf, t);
}

// Copy 16 bytes global -> LDS for this thread; async (no VGPR round-trip) when
// the gfx1250 async builtin is available.
__device__ __forceinline__ void cp16_g2l(const __bf16* g, __bf16* l) {
#if defined(USE_ASYNC_LDS)
    typedef __attribute__((address_space(1))) v4i gv4i_t;   // global-space int4
    typedef __attribute__((address_space(3))) v4i lv4i_t;   // LDS-space int4
    gv4i_t* gp = (gv4i_t*)(unsigned long long)(const void*)g;
    lv4i_t* lp = (lv4i_t*)(unsigned)(unsigned long long)(void*)l;  // LDS offset = addr[31:0]
    __builtin_amdgcn_global_load_async_to_lds_b128(gp, lp, 0, 0);
#else
    *(v8bf*)l = *(const v8bf*)g;
#endif
}

__device__ __forceinline__ void wait_async_done() {
#if defined(USE_ASYNC_LDS)
#if __has_builtin(__builtin_amdgcn_s_wait_asynccnt)
    __builtin_amdgcn_s_wait_asynccnt(0);
#else
    asm volatile("s_wait_asynccnt 0x0" ::: "memory");
#endif
#endif
}

// ---------------------------------------------------------------------------
// Prep: W fp32 -> bf16 in B-fragment layout.
//   Wsw[mat][k32][t][lane][j] = W[k32*32 + (j>>3)*16 + hi*8 + (j&7)][t*16 + n],  lane = hi*16+n
// ---------------------------------------------------------------------------
__global__ void __launch_bounds__(256)
wprep_kernel(const float* __restrict__ Wq, const float* __restrict__ Wk,
             const float* __restrict__ Wv, __bf16* __restrict__ Wsw)
{
    const int idx = blockIdx.x * 256 + threadIdx.x;   // 0 .. 3*WMAT-1
    const int mat = idx >> 16;
    const int e   = idx & (WMAT - 1);
    const int j    = e & 15;
    const int lane = (e >> 4) & 31;
    const int t    = (e >> 9) & 3;
    const int k32  = e >> 11;
    const int n  = lane & 15;
    const int hi = lane >> 4;
    const float* W = (mat == 0) ? Wq : ((mat == 1) ? Wk : Wv);
    Wsw[idx] = (__bf16)W[(size_t)(k32 * 32 + (j >> 3) * 16 + hi * 8 + (j & 7)) * DH + t * 16 + n];
}

// ---------------------------------------------------------------------------
// Kernel 1: QKV projection + RoPE, swizzled bf16 outputs.
// Block = 256 threads = 8 waves; each wave owns one 16-row tile.
// ---------------------------------------------------------------------------
__global__ void __launch_bounds__(256)
qkv_rope_kernel(const float* __restrict__ x,
                const __bf16* __restrict__ Wsw,
                __bf16* __restrict__ Qsw,     // A-frag layout per 16-row tile
                __bf16* __restrict__ Ksw,     // B-frag layout per 16-key tile
                __bf16* __restrict__ Vsw)     // B-frag layout per 32-key tile
{
    const int lane = threadIdx.x & 31;
    const int wave = threadIdx.x >> 5;
    const int n  = lane & 15;
    const int hi = lane >> 4;
    const long tile = (long)blockIdx.x * 8 + wave;   // 16-row tile id
    const long row0 = tile * 16;

    const __bf16* WswQ = Wsw;
    const __bf16* WswK = Wsw + WMAT;
    const __bf16* WswV = Wsw + 2 * WMAT;

    v8f zero = {};
    v8f accQ[4], accK[4], accV[4];
#pragma unroll
    for (int t = 0; t < 4; ++t) { accQ[t] = zero; accK[t] = zero; accV[t] = zero; }

    for (int k32 = 0; k32 < EMB / 32; ++k32) {
        // A fragment: 16x32 tile of x, native f32->bf16 converts (v_cvt_pk_bf16_f32).
        const float* xrow = x + (size_t)(row0 + n) * EMB + k32 * 32 + hi * 8;
        const float4 f0 = *(const float4*)(xrow);
        const float4 f1 = *(const float4*)(xrow + 4);
        const float4 f2 = *(const float4*)(xrow + 16);
        const float4 f3 = *(const float4*)(xrow + 20);
        v16bf a;
        a[0] = (__bf16)f0.x; a[1] = (__bf16)f0.y; a[2] = (__bf16)f0.z; a[3] = (__bf16)f0.w;
        a[4] = (__bf16)f1.x; a[5] = (__bf16)f1.y; a[6] = (__bf16)f1.z; a[7] = (__bf16)f1.w;
        a[8] = (__bf16)f2.x; a[9] = (__bf16)f2.y; a[10] = (__bf16)f2.z; a[11] = (__bf16)f2.w;
        a[12] = (__bf16)f3.x; a[13] = (__bf16)f3.y; a[14] = (__bf16)f3.z; a[15] = (__bf16)f3.w;

#pragma unroll
        for (int t = 0; t < 4; ++t) {
            const size_t woff = (size_t)((k32 * 4 + t) * 32 + lane) * 16;
            const v16bf bq = load_frag(WswQ + woff);
            const v16bf bk = load_frag(WswK + woff);
            const v16bf bv = load_frag(WswV + woff);
            accQ[t] = __builtin_amdgcn_wmma_f32_16x16x32_bf16(
                false, a, false, bq, (short)0, accQ[t], false, false);
            accK[t] = __builtin_amdgcn_wmma_f32_16x16x32_bf16(
                false, a, false, bk, (short)0, accK[t], false, false);
            accV[t] = __builtin_amdgcn_wmma_f32_16x16x32_bf16(
                false, a, false, bv, (short)0, accV[t], false, false);
        }
    }

    // RoPE + swizzled stores. C/D layout: VGPR v -> row m = v+8*hi, lane&15 = col.
    const int s_base = (int)(row0 % SEQ);
    const float LOG2_THETA = 13.287712379549449f;   // log2(10000)

#pragma unroll
    for (int t = 0; t < 4; ++t) {
        const int d = t * 16 + n;
        const int p = d >> 1;
        const bool odd = (d & 1) != 0;
        const float inv_freq = exp2f(-(2.0f * (float)p / (float)DH) * LOG2_THETA);

        // Q destination swizzle (A-fragment layout)
        const int q_kq = d >> 5;
        const int q_hi = (d >> 3) & 1;
        const int q_j  = ((d >> 4) & 1) * 8 + (d & 7);
        // K destination swizzle (B-fragment layout, key tile = this row tile)
        const int k_kq = d >> 5;
        const int k_hi = (d >> 3) & 1;
        const int k_j  = ((d >> 4) & 1) * 8 + (d & 7);

#pragma unroll
        for (int v = 0; v < 8; ++v) {
            const int m = v + 8 * hi;
            const float pos = (float)(s_base + m);
            const float th = pos * inv_freq;
            const float sn = sinf(th), cs = cosf(th);

            // Q with RoPE
            float aq = accQ[t][v];
            float oq = __shfl_xor(aq, 1, 32);
            float q1 = odd ? oq : aq;
            float q2 = odd ? aq : oq;
            float rq = odd ? (q1 * sn + q2 * cs) : (q1 * cs - q2 * sn);
            Qsw[((size_t)(tile * 2 + q_kq) * 32 + (q_hi * 16 + m)) * 16 + q_j] = (__bf16)rq;

            // K with RoPE
            float ak = accK[t][v];
            float ok = __shfl_xor(ak, 1, 32);
            float k1 = odd ? ok : ak;
            float k2 = odd ? ak : ok;
            float rk = odd ? (k1 * sn + k2 * cs) : (k1 * cs - k2 * sn);
            Ksw[((size_t)(tile * 2 + k_kq) * 32 + (k_hi * 16 + m)) * 16 + k_j] = (__bf16)rk;

            // V (no rope): B-frag per 32-key tile: hi' = tile&1, j = m
            Vsw[(((size_t)(tile >> 1) * 4 + t) * 32 + ((tile & 1) * 16 + n)) * 16 + m]
                = (__bf16)accV[t][v];
        }
    }
}

// ---------------------------------------------------------------------------
// Online-softmax step update. MASKED selects the causal-boundary variant;
// interior steps (ks+31 <= s0) take the mask-free path.
// ---------------------------------------------------------------------------
template <bool MASKED>
__device__ __forceinline__ void softmax_update(
    const v8f& sc0, const v8f& sc1, int ks, int s0, int n, int hi,
    float* mi, float* li, v8f* o, __bf16* pb)
{
#pragma unroll
    for (int v = 0; v < 8; ++v) {
        const int m = v + 8 * hi;
        float s1 = sc0[v] * 0.125f;
        float s2 = sc1[v] * 0.125f;
        if (MASKED) {
            const int srow = s0 + m;
            if (ks + n      > srow) s1 = -3.0e38f;
            if (ks + 16 + n > srow) s2 = -3.0e38f;
        }

        float mcur = fmaxf(s1, s2);
        mcur = fmaxf(mcur, __shfl_xor(mcur, 1, 32));
        mcur = fmaxf(mcur, __shfl_xor(mcur, 2, 32));
        mcur = fmaxf(mcur, __shfl_xor(mcur, 4, 32));
        mcur = fmaxf(mcur, __shfl_xor(mcur, 8, 32));

        const float mnew  = fmaxf(mi[v], mcur);
        const float alpha = __expf(mi[v] - mnew);
        const float p1 = __expf(s1 - mnew);
        const float p2 = __expf(s2 - mnew);

        float rs = p1 + p2;
        rs += __shfl_xor(rs, 1, 32);
        rs += __shfl_xor(rs, 2, 32);
        rs += __shfl_xor(rs, 4, 32);
        rs += __shfl_xor(rs, 8, 32);

        li[v] = li[v] * alpha + rs;
        mi[v] = mnew;

        pb[m * 32 + n]      = (__bf16)p1;
        pb[m * 32 + 16 + n] = (__bf16)p2;

        o[0][v] *= alpha; o[1][v] *= alpha; o[2][v] *= alpha; o[3][v] *= alpha;
    }
}

// ---------------------------------------------------------------------------
// Kernel 2: causal flash attention with async LDS staging of K/V.
// Block = 128 threads = 4 waves covering 64 query rows; each wave owns one
// 16-query tile. K/V 32-key step tiles are contiguous 4KB regions in the
// swizzled layouts. Grid = NROW/64 = 256 blocks.
// ---------------------------------------------------------------------------
__global__ void __launch_bounds__(128)
attn_kernel(const __bf16* __restrict__ Qsw,
            const __bf16* __restrict__ Ksw,
            const __bf16* __restrict__ Vsw,
            float* __restrict__ out)
{
    __shared__ __attribute__((aligned(16))) __bf16 kbuf[2][2048];  // 2 x 4KB
    __shared__ __attribute__((aligned(16))) __bf16 vbuf[2][2048];  // 2 x 4KB
    __shared__ __attribute__((aligned(16))) __bf16 pbuf[4][16 * 32];

    const int tid  = threadIdx.x;
    const int lane = tid & 31;
    const int wave = tid >> 5;
    const int n  = lane & 15;
    const int hi = lane >> 4;

    // Block -> (batch, 64-row group); permute groups so heavy (late-sequence)
    // and light blocks interleave in launch order (causal load balance).
    const int  batch = blockIdx.x >> 6;          // 64 groups per batch
    const int  sg    = blockIdx.x & 63;
    const int  sgp   = (sg * 37) & 63;           // odd multiplier: permutation
    const long brow0 = (long)batch * SEQ;
    const int  s0b   = sgp * 64;                 // block's first seq position
    const int  s0    = s0b + wave * 16;          // this wave's first query row
    const long row0  = brow0 + s0;
    const long qtile = row0 >> 4;

    // Q A-fragments: two coalesced 32B loads each.
    v16bf aQ[2];
#pragma unroll
    for (int kq = 0; kq < 2; ++kq)
        aQ[kq] = load_frag(Qsw + ((size_t)(qtile * 2 + kq) * 32 + lane) * 16);

    v8f zero = {};
    v8f o[4];
#pragma unroll
    for (int t = 0; t < 4; ++t) o[t] = zero;
    float mi[8], li[8];
#pragma unroll
    for (int v = 0; v < 8; ++v) { mi[v] = -3.0e38f; li[v] = 0.0f; }

    const int nsteps = (s0b + 64) >> 5;          // block-uniform trip count

    // ---- stage step 0 (K and V step tiles both start at (key_row)*64 elems)
    cp16_g2l(Ksw + (size_t)brow0 * 64 + tid * 32,      &kbuf[0][tid * 32]);
    cp16_g2l(Ksw + (size_t)brow0 * 64 + tid * 32 + 16, &kbuf[0][tid * 32 + 16]);
    cp16_g2l(Vsw + (size_t)brow0 * 64 + tid * 32,      &vbuf[0][tid * 32]);
    cp16_g2l(Vsw + (size_t)brow0 * 64 + tid * 32 + 16, &vbuf[0][tid * 32 + 16]);
    wait_async_done();
    __syncthreads();

    for (int step = 0; step < nsteps; ++step) {
        const int  cur  = step & 1;
        const int  ks   = step * 32;
        const long krow = brow0 + ks;

        // issue next step's staging (async: lands in LDS behind the compute)
        const bool more = (step + 1 < nsteps);
        if (more) {
            const __bf16* gk = Ksw + (size_t)(krow + 32) * 64 + tid * 32;
            const __bf16* gv = Vsw + (size_t)(krow + 32) * 64 + tid * 32;
            cp16_g2l(gk,      &kbuf[cur ^ 1][tid * 32]);
            cp16_g2l(gk + 16, &kbuf[cur ^ 1][tid * 32 + 16]);
            cp16_g2l(gv,      &vbuf[cur ^ 1][tid * 32]);
            cp16_g2l(gv + 16, &vbuf[cur ^ 1][tid * 32 + 16]);
        }
        if (step + 2 < nsteps) {  // warm L2 two steps ahead (global_prefetch path)
            __builtin_prefetch(Ksw + (size_t)(krow + 64) * 64 + tid * 16, 0, 1);
            __builtin_prefetch(Vsw + (size_t)(krow + 64) * 64 + tid * 16, 0, 1);
        }

        // wave-uniform: skip compute entirely for fully-masked steps
        if (ks <= s0 + 15) {
            // ---- scores: two 16-key N-tiles, K=DH via 2 WMMAs each (K from LDS)
            v8f sc[2];
#pragma unroll
            for (int t = 0; t < 2; ++t) {
                v8f s = zero;
#pragma unroll
                for (int kq = 0; kq < 2; ++kq) {
                    const v16bf bk = load_frag(&kbuf[cur][((t * 2 + kq) * 32 + lane) * 16]);
                    s = __builtin_amdgcn_wmma_f32_16x16x32_bf16(
                        false, aQ[kq], false, bk, (short)0, s, false, false);
                }
                sc[t] = s;
            }

            // ---- online softmax: mask-free on interior steps (wave-uniform)
            if (ks + 31 <= s0)
                softmax_update<false>(sc[0], sc[1], ks, s0, n, hi, mi, li, o, pbuf[wave]);
            else
                softmax_update<true>(sc[0], sc[1], ks, s0, n, hi, mi, li, o, pbuf[wave]);

            // ---- reload P as an A fragment (two ds_load_b128)
            const v16bf aP = load_frag(&pbuf[wave][n * 32 + hi * 16]);

            // ---- O += P @ V : 4 N-tiles over DH, K = 32 keys (V from LDS)
#pragma unroll
            for (int t = 0; t < 4; ++t) {
                const v16bf bv = load_frag(&vbuf[cur][(t * 32 + lane) * 16]);
                o[t] = __builtin_amdgcn_wmma_f32_16x16x32_bf16(
                    false, aP, false, bv, (short)0, o[t], false, false);
            }
        }

        wait_async_done();     // own async stores to LDS complete
        __syncthreads();       // everyone's staging visible, reads of cur done
    }

    // ---- normalize and store fp32 output (coalesced b32 across lanes)
#pragma unroll
    for (int t = 0; t < 4; ++t) {
        const int d = t * 16 + n;
#pragma unroll
        for (int v = 0; v < 8; ++v) {
            const int m = v + 8 * hi;
            out[(size_t)(row0 + m) * DH + d] = o[t][v] * (1.0f / li[v]);
        }
    }
}

// ---------------------------------------------------------------------------
// Launcher. Inputs: x, mask(unused; causal implemented directly), W_Q, W_K, W_V.
// Workspace: Wsw bf16 (384KB) | Qsw (2MB) | Ksw (2MB) | Vsw (2MB).
// ---------------------------------------------------------------------------
extern "C" void kernel_launch(void* const* d_in, const int* in_sizes, int n_in,
                              void* d_out, int out_size, void* d_ws, size_t ws_size,
                              hipStream_t stream) {
    const float* x  = (const float*)d_in[0];
    const float* Wq = (const float*)d_in[2];
    const float* Wk = (const float*)d_in[3];
    const float* Wv = (const float*)d_in[4];
    float* out = (float*)d_out;

    __bf16* Wsw = (__bf16*)d_ws;
    __bf16* Qsw = Wsw + (size_t)3 * WMAT;
    __bf16* Ksw = Qsw + (size_t)NROW * DH;
    __bf16* Vsw = Ksw + (size_t)NROW * DH;

    wprep_kernel<<<3 * WMAT / 256, 256, 0, stream>>>(Wq, Wk, Wv, Wsw);
    qkv_rope_kernel<<<NROW / 16 / 8, 256, 0, stream>>>(x, Wsw, Qsw, Ksw, Vsw);
    attn_kernel<<<NROW / 64, 128, 0, stream>>>(Qsw, Ksw, Vsw, out);
}
